// Self_Attention_3332894622394
// MI455X (gfx1250) — matrix operand order repaired and checked by reading, not compiled, not run on hardware
//
#include <hip/hip_runtime.h>
#include <math.h>

// Multi-key attention (N=8, K=8, C=64, H=W=32 -> L=1024), fp32 in/out.
// Flash-attention: one wave owns a 16-row query block; block = 8 waves = 128 rows.
// f16 WMMA (v_wmma_f32_16x16x32_f16) with f32 accumulation.
//  - Q pre-scaled by log2(e) at staging -> exp via single v_exp (base-2 domain)
//  - softmax denominator accumulated on the XDL pipe via a ones-column WMMA
//  - K/V staged in LDS as f16 (one cvt per element), register-pipelined tiles
// Layouts are channel-major: q[n][c][l], k/v[n][k][c][l], out[n][k][c][l].

#define N_BATCH 8
#define K_KEYS  8
#define C_DIM   64
#define L_SEQ   1024
#define BQ      128               // query rows per block (8 waves x 16)
#define BP      64                // key positions per iteration
#define N_ITERS (L_SEQ / BP)      // 16
#define QTILES  (L_SEQ / BQ)      // 8
#define LOG2E   1.4426950408889634f

typedef __attribute__((ext_vector_type(16))) _Float16 v16h;
typedef __attribute__((ext_vector_type(8)))  _Float16 h8;
typedef __attribute__((ext_vector_type(4)))  _Float16 h4;
typedef __attribute__((ext_vector_type(8)))  float    v8f;
typedef __attribute__((ext_vector_type(4)))  float    f4;

// B-fragment from 16 *consecutive* f16 in LDS (element e <-> within-WMMA K = h*16+e;
// the caller bakes h*16 into the address). 32B = two 16B ds loads.
__device__ __forceinline__ v16h b_frag_f16(const _Float16* p) {
  h8 lo = *(const h8*)p;
  h8 hi = *(const h8*)(p + 8);
  v16h f;
#pragma unroll
  for (int e = 0; e < 8; ++e) { f[e] = lo[e]; f[8 + e] = hi[e]; }
  return f;
}

// A-fragment: two 8-half (16B) runs, contiguous.
// Element e <-> K=h*16+e; caller bakes h*16 into address.
__device__ __forceinline__ v16h a_frag_f16(const _Float16* p) {
  h8 lo = *(const h8*)p;
  h8 hi = *(const h8*)(p + 8);
  v16h f;
#pragma unroll
  for (int e = 0; e < 8; ++e) { f[e] = lo[e]; f[8 + e] = hi[e]; }
  return f;
}

__global__ __launch_bounds__(256)
void mk_attn_kernel(const float* __restrict__ query,
                    const float* __restrict__ key,
                    const float* __restrict__ value,
                    float* __restrict__ out) {
  __shared__ _Float16 Qs[BQ * C_DIM];        // [q][c]  16 KB (pre-scaled by log2e)
  __shared__ _Float16 Ks[2][BP * C_DIM];     // [p][c]  (transposed) 2x8 KB
  __shared__ _Float16 Vs[2][C_DIM * BP];     // [c][p]  2x8 KB
  __shared__ _Float16 Ps[8][16 * BP];        // per-wave P scratch 8x2 KB

  const int tid  = (int)threadIdx.x;
  const int wave = tid >> 5;
  const int lane = tid & 31;
  const int h    = lane >> 4;   // lane half (wave32)
  const int ln   = lane & 15;

  const int qt   = (int)blockIdx.x & (QTILES - 1);
  const int pair = (int)blockIdx.x / QTILES;      // n*K_KEYS + k
  const float* qG = query + (pair >> 3) * (C_DIM * L_SEQ);
  const float* kG = key   + pair * (C_DIM * L_SEQ);
  const float* vG = value + pair * (C_DIM * L_SEQ);
  float*       oG = out   + pair * (C_DIM * L_SEQ);
  const int q0 = qt * BQ;

  // ---- stage Q tile: global [c][q] -> Qs[q][c] f16, scaled by log2e ----
#pragma unroll
  for (int i = 0; i < (BQ * C_DIM) / 256; ++i) {
    int idx = tid + i * 256;
    int c = idx >> 7, ql = idx & (BQ - 1);
    Qs[ql * C_DIM + c] = (_Float16)(qG[c * L_SEQ + q0 + ql] * LOG2E);
  }

  // ---- stage first K/V tile (f32 -> f16, once per element) ----
#pragma unroll
  for (int i = 0; i < 16; ++i) {             // K: transpose-scatter, 2B stores
    int idx = tid + i * 256;
    int c = idx >> 6, p = idx & 63;
    Ks[0][p * C_DIM + c] = (_Float16)kG[c * L_SEQ + p];
  }
#pragma unroll
  for (int i = 0; i < 4; ++i) {              // V: straight 4-wide chunks
    int idx = tid + i * 256;
    int c = idx >> 4, p = (idx & 15) * 4;
    f4 v = *(const f4*)(vG + c * L_SEQ + p);
    h4 t;
#pragma unroll
    for (int e = 0; e < 4; ++e) t[e] = (_Float16)v[e];
    *(h4*)&Vs[0][idx * 4] = t;
  }
  __syncthreads();

  // ---- Q A-fragments for both C halves (row r = ln) ----
  v16h aQ[2];
#pragma unroll
  for (int a = 0; a < 2; ++a)
    aQ[a] = a_frag_f16(&Qs[(wave * 16 + ln) * C_DIM + a * 32 + h * 16]);

  // ---- ones-column B fragment: B[k][n] = (n==0), zero LDS cost ----
  v16h bOnes;
  {
    _Float16 one = (_Float16)((ln == 0) ? 1.0f : 0.0f);
#pragma unroll
    for (int e = 0; e < 16; ++e) bOnes[e] = one;
  }

  // ---- flash-attention state (rows m = 8*h + j for this lane) ----
  v8f Oc[4], Osum;
  float rmax[8];
#pragma unroll
  for (int j = 0; j < 8; ++j) {
    rmax[j] = -__builtin_inff();
    Osum[j] = 0.0f;
#pragma unroll
    for (int ct = 0; ct < 4; ++ct) Oc[ct][j] = 0.0f;
  }

  _Float16* myP = Ps[wave];

  for (int it = 0; it < N_ITERS; ++it) {
    const int cur = it & 1;
    const bool haveNext = (it + 1 < N_ITERS);

    // ---- issue global loads for tile it+1 into registers (hide under compute)
    float kreg[16];
    f4    vreg[4];
    if (haveNext) {
      const int pN = (it + 1) * BP;
#pragma unroll
      for (int i = 0; i < 16; ++i) {
        int idx = tid + i * 256;
        int c = idx >> 6, p = idx & 63;
        kreg[i] = kG[c * L_SEQ + pN + p];
      }
#pragma unroll
      for (int i = 0; i < 4; ++i) {
        int idx = tid + i * 256;
        int c = idx >> 4, p = (idx & 15) * 4;
        vreg[i] = *(const f4*)(vG + c * L_SEQ + pN + p);
      }
    }
    if (it + 2 < N_ITERS && tid < C_DIM) {   // global_prefetch_b8 two tiles out
      __builtin_prefetch(kG + tid * L_SEQ + (it + 2) * BP, 0, 1);
      __builtin_prefetch(vG + tid * L_SEQ + (it + 2) * BP, 0, 1);
    }

    // ---- S = (log2e*Q) K^T : four 16x16 p-subtiles, K-dim = C = 64 ----
    v8f S[4];
#pragma unroll
    for (int pb = 0; pb < 4; ++pb) {
      v16h b0 = b_frag_f16(&Ks[cur][(pb * 16 + ln) * C_DIM + h * 16]);
      v16h b1 = b_frag_f16(&Ks[cur][(pb * 16 + ln) * C_DIM + 32 + h * 16]);
      v8f s;
#pragma unroll
      for (int j = 0; j < 8; ++j) s[j] = 0.0f;
      s = __builtin_amdgcn_wmma_f32_16x16x32_f16(false, aQ[0], false, b0, (short)0, s, false, false);
      s = __builtin_amdgcn_wmma_f32_16x16x32_f16(false, aQ[1], false, b1, (short)0, s, false, false);
      S[pb] = s;
    }

    // ---- online softmax, base-2 domain (row m = 8h+j; 16-lane half reductions)
    float mn[8], sc[8];
#pragma unroll
    for (int j = 0; j < 8; ++j) {
      float bm = fmaxf(fmaxf(S[0][j], S[1][j]), fmaxf(S[2][j], S[3][j]));
      bm = fmaxf(bm, __shfl_xor(bm, 1, 32));
      bm = fmaxf(bm, __shfl_xor(bm, 2, 32));
      bm = fmaxf(bm, __shfl_xor(bm, 4, 32));
      bm = fmaxf(bm, __shfl_xor(bm, 8, 32));
      mn[j] = fmaxf(rmax[j], bm);
      sc[j] = exp2f(rmax[j] - mn[j]);
      rmax[j] = mn[j];
    }
    // exp2 and emit P (f16) to per-wave LDS scratch (sum comes from WMMA)
#pragma unroll
    for (int pb = 0; pb < 4; ++pb)
#pragma unroll
      for (int j = 0; j < 8; ++j) {
        float e = exp2f(S[pb][j] - mn[j]);
        myP[(8 * h + j) * BP + pb * 16 + ln] = (_Float16)e;
      }
    // rescale running accumulators (including the sum column)
#pragma unroll
    for (int j = 0; j < 8; ++j) {
      Osum[j] *= sc[j];
#pragma unroll
      for (int ct = 0; ct < 4; ++ct) Oc[ct][j] *= sc[j];
    }

    __builtin_amdgcn_wave_barrier();          // per-wave DS ops are in-order
    v16h aP0 = a_frag_f16(&myP[ln * BP + h * 16]);
    v16h aP1 = a_frag_f16(&myP[ln * BP + 32 + h * 16]);
    __builtin_amdgcn_wave_barrier();

    // ---- O += P V, and Osum += P * ones (denominator on the XDL pipe) ----
#pragma unroll
    for (int ct = 0; ct < 4; ++ct) {
      v16h b0 = b_frag_f16(&Vs[cur][(ct * 16 + ln) * BP + h * 16]);
      v16h b1 = b_frag_f16(&Vs[cur][(ct * 16 + ln) * BP + 32 + h * 16]);
      Oc[ct] = __builtin_amdgcn_wmma_f32_16x16x32_f16(false, aP0, false, b0, (short)0, Oc[ct], false, false);
      Oc[ct] = __builtin_amdgcn_wmma_f32_16x16x32_f16(false, aP1, false, b1, (short)0, Oc[ct], false, false);
    }
    Osum = __builtin_amdgcn_wmma_f32_16x16x32_f16(false, aP0, false, bOnes, (short)0, Osum, false, false);
    Osum = __builtin_amdgcn_wmma_f32_16x16x32_f16(false, aP1, false, bOnes, (short)0, Osum, false, false);

    // ---- flush staged tile it+1 into the other LDS buffer (f32 -> f16) ----
    if (haveNext) {
      const int nxt = cur ^ 1;
#pragma unroll
      for (int i = 0; i < 16; ++i) {
        int idx = tid + i * 256;
        int c = idx >> 6, p = idx & 63;
        Ks[nxt][p * C_DIM + c] = (_Float16)kreg[i];
      }
#pragma unroll
      for (int i = 0; i < 4; ++i) {
        int idx = tid + i * 256;
        h4 t;
#pragma unroll
        for (int e = 0; e < 4; ++e) t[e] = (_Float16)vreg[i][e];
        *(h4*)&Vs[nxt][idx * 4] = t;
      }
    }
    __syncthreads();     // everyone done reading cur + nxt fully staged
  }

  // ---- recover row sums from Osum column 0 (lane n==0 of own half) ----
  float rinv[8];
#pragma unroll
  for (int j = 0; j < 8; ++j)
    rinv[j] = 1.0f / __shfl(Osum[j], h * 16, 32);

  // ---- normalize and store: out[c][q], 8 contiguous floats per lane/tile ----
  const int qb = q0 + wave * 16 + 8 * h;
#pragma unroll
  for (int ct = 0; ct < 4; ++ct) {
    int c = ct * 16 + ln;
    float* dst = oG + c * L_SEQ + qb;
    f4 lo, hi;
#pragma unroll
    for (int j = 0; j < 4; ++j) {
      lo[j] = Oc[ct][j] * rinv[j];
      hi[j] = Oc[ct][4 + j] * rinv[4 + j];
    }
    *(f4*)dst = lo;
    *(f4*)(dst + 4) = hi;
  }
}

extern "C" void kernel_launch(void* const* d_in, const int* in_sizes, int n_in,
                              void* d_out, int out_size, void* d_ws, size_t ws_size,
                              hipStream_t stream) {
  (void)in_sizes; (void)n_in; (void)d_ws; (void)ws_size; (void)out_size;
  const float* q = (const float*)d_in[0];
  const float* k = (const float*)d_in[1];
  const float* v = (const float*)d_in[2];
  float* o = (float*)d_out;
  dim3 grid(N_BATCH * K_KEYS * QTILES);   // 512 blocks
  dim3 block(256);                        // 8 wave32 waves
  mk_attn_kernel<<<grid, block, 0, stream>>>(q, k, v, o);
}
